// CustomLSTM_With_Peephole_40845138985284
// MI455X (gfx1250) — matrix operand order repaired and checked by reading, not compiled, not run on hardware
//
#include <hip/hip_runtime.h>
#include <stdint.h>
#include <math.h>

#define Bn 256
#define Tn 512
#define In 128
#define Hn 256
#define HB_STRIDE 264    // bf16 elems per hbuf row (256 + 8 pad)  -> conflict-free ds_load_b128
#define XS_STRIDE 132    // fp32 elems per x staging row (128 + 4) -> written by TDM pad feature
#define XB_STRIDE 136    // bf16 elems per x bf16 row (128 + 8)    -> conflict-free ds_load_b128

typedef __bf16 bf16_t;
typedef __attribute__((ext_vector_type(16))) bf16_t v16bf;
typedef __attribute__((ext_vector_type(8)))  bf16_t v8bf;
typedef __attribute__((ext_vector_type(4)))  bf16_t v4bf;
typedef __attribute__((ext_vector_type(8)))  float  v8f;
typedef __attribute__((ext_vector_type(4)))  float  v4f;
typedef __attribute__((ext_vector_type(4)))  unsigned int v4u;
typedef __attribute__((ext_vector_type(8)))  int    v8i;
typedef __attribute__((ext_vector_type(4)))  int    v4i;

// ---------------------------------------------------------------------------
// One-time weight prep: fp32 [K,N] row-major -> bf16 [N,K] (transposed) so the
// WMMA B-operand (col N fixed per lane, K contiguous) is a single 32B load.
// ---------------------------------------------------------------------------
__global__ void cvt_transpose_kernel(const float* __restrict__ src,
                                     bf16_t* __restrict__ dst,
                                     int K, int N) {
  int idx = blockIdx.x * blockDim.x + threadIdx.x;
  if (idx >= K * N) return;
  int k = idx / N;
  int n = idx - k * N;
  dst[(size_t)n * K + k] = (bf16_t)src[idx];
}

__device__ __forceinline__ float sigmoidf_(float x) {
  return 1.0f / (1.0f + __expf(-x));
}
__device__ __forceinline__ float tanhf_fast(float x) {
  float e = __expf(2.0f * x);            // v_exp_f32 fast path
  return 1.0f - 2.0f / (e + 1.0f);       // exact limits at +/-inf
}

__device__ __forceinline__ v8f splat8(float v) {
  v8f r;
#pragma unroll
  for (int i = 0; i < 8; ++i) r[i] = v;
  return r;
}

__device__ __forceinline__ v8f wmma_bf16(v16bf a, v16bf b, v8f c) {
  return __builtin_amdgcn_wmma_f32_16x16x32_bf16(false, a, false, b,
                                                 (short)0, c, false, false);
}

// A-operand from LDS bf16, contiguous-K layout: [0..15] = K k..k+15.
__device__ __forceinline__ v16bf load_a_lds(const bf16_t* p) {
  v8bf lo = *(const v8bf*)(p);
  v8bf hi = *(const v8bf*)(p + 8);
  return __builtin_shufflevector(lo, hi, 0, 1, 2, 3, 4, 5, 6, 7,
                                 8, 9, 10, 11, 12, 13, 14, 15);
}

// ---------------------------------------------------------------------------
// TDM: DMA one x tile [16 rows x 128 fp32] (row stride T*I) into LDS, with the
// D# pad feature inserting 4 DWORDs after every 128 DWORDs so the tile lands
// directly in the bank-conflict-free XS_STRIDE layout.
// ---------------------------------------------------------------------------
__device__ __forceinline__ void tdm_load_x_tile(const float* gsrc,
                                                unsigned int lds_byte_addr) {
  uint64_t ga = (uint64_t)(uintptr_t)gsrc;
  v4u g0;
  g0[0] = 1u;                                            // count=1 (valid user D#)
  g0[1] = lds_byte_addr;                                 // lds_addr
  g0[2] = (unsigned int)ga;                              // global_addr[31:0]
  g0[3] = (unsigned int)((ga >> 32) & 0x01FFFFFFu)       // global_addr[56:32]
          | 0x80000000u;                                 // type=2 ("image")
  v8i g1;
  g1[0] = (int)((2u << 16)      // data_size = 4 bytes
              | (1u << 20)      // pad_enable
              | (6u << 22)      // pad_interval: 128 DWORDs (one row)
              | (3u << 25));    // pad_amount: 4 DWORDs
  g1[1] = (int)((unsigned int)(In & 0xFFFF) << 16);      // tensor_dim0[15:0]
  g1[2] = (int)(((unsigned int)In >> 16) | (16u << 16)); // tensor_dim0 hi, tensor_dim1=16
  g1[3] = (int)((unsigned int)In << 16);                 // tile_dim0 = 128
  g1[4] = (int)16;                                       // tile_dim1 = 16, tile_dim2 = 0
  g1[5] = (int)(Tn * In);                                // tensor_dim0_stride = 65536
  g1[6] = 0;
  g1[7] = 0;
  v4i gz = {0, 0, 0, 0};
#if defined(__clang_major__) && (__clang_major__ >= 23)
  v8i g4 = {0, 0, 0, 0, 0, 0, 0, 0};
  __builtin_amdgcn_tensor_load_to_lds(g0, g1, gz, gz, g4, 0);
#else
  __builtin_amdgcn_tensor_load_to_lds(g0, g1, gz, gz, 0);
#endif
}

// ---------------------------------------------------------------------------
// Persistent recurrent kernel: 16 WGs (one per 16-row batch tile), 16 waves
// each (one per 16-wide H slice). c/h/gates live in the WMMA C/D register
// layout; h round-trips through LDS bf16; x streams in via TDM and is
// converted to bf16 once per step, cooperatively, in the barrier window.
// ---------------------------------------------------------------------------
__launch_bounds__(512, 1)
__global__ void lstm_peephole_wmma_kernel(
    const float* __restrict__ x,
    const bf16_t* __restrict__ Ut,   // [4][Hn][In]  transposed bf16 (f,i,c,o)
    const bf16_t* __restrict__ Wt,   // [4][Hn][Hn]  transposed bf16 (f,i,c,o)
    const float* __restrict__ Vf, const float* __restrict__ Vi,
    const float* __restrict__ Vo,
    const float* __restrict__ bfp, const float* __restrict__ bip,
    const float* __restrict__ bcp, const float* __restrict__ bop,
    float* __restrict__ out) {
  __shared__ bf16_t hbuf[16 * HB_STRIDE];     // h state, bf16
  __shared__ float  xstage[16 * XS_STRIDE];   // TDM fp32 landing pad
  __shared__ bf16_t xb16[16 * XB_STRIDE];     // converted x tile, bf16

  const int lane = threadIdx.x & 31;   // wave32
  const int wave = threadIdx.x >> 5;   // 0..15 -> N tile index
  const int half = lane >> 4;
  const int m    = lane & 15;
  const int n    = wave * 16 + m;      // hidden unit owned by this lane
  const int b0   = blockIdx.x * 16;    // batch tile base row

  const float vf  = Vf[n],  vi  = Vi[n],  vo  = Vo[n];
  const float bfv = bfp[n], biv = bip[n], bcv = bcp[n], bov = bop[n];

  // B-operand row pointers: contiguous-K per lane, half selects K+16 block.
  // Provenance kept intact (no asm on pointers) -> global_load_b128.
  const bf16_t* pu0 = Ut + (size_t)0 * Hn * In + (size_t)n * In + half * 16;
  const bf16_t* pu1 = Ut + (size_t)1 * Hn * In + (size_t)n * In + half * 16;
  const bf16_t* pu2 = Ut + (size_t)2 * Hn * In + (size_t)n * In + half * 16;
  const bf16_t* pu3 = Ut + (size_t)3 * Hn * In + (size_t)n * In + half * 16;
  const bf16_t* pw0 = Wt + (size_t)0 * Hn * Hn + (size_t)n * Hn + half * 16;
  const bf16_t* pw1 = Wt + (size_t)1 * Hn * Hn + (size_t)n * Hn + half * 16;
  const bf16_t* pw2 = Wt + (size_t)2 * Hn * Hn + (size_t)n * Hn + half * 16;
  const bf16_t* pw3 = Wt + (size_t)3 * Hn * Hn + (size_t)n * Hn + half * 16;

  const bf16_t* hrow = hbuf + m * HB_STRIDE + half * 16;
  const bf16_t* xrow = xb16 + m * XB_STRIDE + half * 16;

  // Cooperative x fp32->bf16 conversion mapping: 512 threads x 4 floats.
  const int cr = threadIdx.x >> 5;          // row 0..15
  const int cc = (threadIdx.x & 31) * 4;    // col 0..124

  // Prime: TDM x_0 (wave 0 only; uniform operands, EXEC ignored by TDM).
  if (wave == 0)
    tdm_load_x_tile(x + (size_t)b0 * Tn * In,
                    (unsigned int)(uintptr_t)&xstage[0]);

  // h(0) = 0
  for (int i = threadIdx.x; i < 16 * HB_STRIDE; i += (int)blockDim.x)
    hbuf[i] = (bf16_t)0.0f;
  if (wave == 0) __builtin_amdgcn_s_wait_tensorcnt(0);
  __syncthreads();
  {
    v4f q = *(const v4f*)&xstage[cr * XS_STRIDE + cc];
    v4bf b;
#pragma unroll
    for (int i = 0; i < 4; ++i) b[i] = (bf16_t)q[i];
    *(v4bf*)&xb16[cr * XB_STRIDE + cc] = b;
  }
  __syncthreads();

  v8f c = splat8(0.0f);
  v8f h = splat8(0.0f);
  int zoff = 0;   // opaque 0: keeps weight loads in-loop without losing AS info

  for (int t = 0; t < Tn; ++t) {
    // Kick off DMA of x_{t+1}; overlaps all compute; staged reads finished at
    // the second barrier of the previous iteration.
    if (wave == 0 && (t + 1) < Tn)
      tdm_load_x_tile(x + ((size_t)b0 * Tn + (t + 1)) * In,
                      (unsigned int)(uintptr_t)&xstage[0]);

    asm volatile("" : "+v"(zoff));   // defeat LICM (addresses loop-variant)

    v8f af = splat8(bfv), ai = splat8(biv), ac = splat8(bcv), ao = splat8(bov);

    // gates += x_t @ U_*   (K = 128 -> 4 k-tiles), bf16 A from LDS
#pragma unroll
    for (int kt = 0; kt < In / 32; ++kt) {
      v16bf a = load_a_lds(xrow + kt * 32);
      af = wmma_bf16(a, *(const v16bf*)(pu0 + kt * 32 + zoff), af);
      ai = wmma_bf16(a, *(const v16bf*)(pu1 + kt * 32 + zoff), ai);
      ac = wmma_bf16(a, *(const v16bf*)(pu2 + kt * 32 + zoff), ac);
      ao = wmma_bf16(a, *(const v16bf*)(pu3 + kt * 32 + zoff), ao);
    }
    // gates += h_{t-1} @ W_*   (K = 256 -> 8 k-tiles)
#pragma unroll
    for (int kt = 0; kt < Hn / 32; ++kt) {
      v16bf a = load_a_lds(hrow + kt * 32);
      af = wmma_bf16(a, *(const v16bf*)(pw0 + kt * 32 + zoff), af);
      ai = wmma_bf16(a, *(const v16bf*)(pw1 + kt * 32 + zoff), ai);
      ac = wmma_bf16(a, *(const v16bf*)(pw2 + kt * 32 + zoff), ac);
      ao = wmma_bf16(a, *(const v16bf*)(pw3 + kt * 32 + zoff), ao);
    }

    // Elementwise gate math in fp32 on the C/D register layout.
#pragma unroll
    for (int r = 0; r < 8; ++r) {
      float cc2 = c[r];
      float fg = sigmoidf_(af[r] + vf * cc2);
      float ig = sigmoidf_(ai[r] + vi * cc2);
      float cn = fg * cc2 + ig * tanhf_fast(ac[r]);
      float og = sigmoidf_(ao[r] + vo * cn);   // peephole on NEW cell
      c[r] = cn;
      h[r] = og * tanhf_fast(cn);
    }

    // hidden_seq[b][t][n]; C/D layout: row M = r + 8*half, col n.
#pragma unroll
    for (int r = 0; r < 8; ++r)
      out[(((size_t)(b0 + r + 8 * half)) * Tn + t) * Hn + n] = h[r];

    // x_{t+1} DMA must be complete before the barrier publishes it.
    if (wave == 0) __builtin_amdgcn_s_wait_tensorcnt(0);
    __syncthreads();   // old h / x_t reads done + staged x_{t+1} visible

    // Barrier window: publish new h (bf16) and convert x_{t+1} to bf16.
#pragma unroll
    for (int r = 0; r < 8; ++r)
      hbuf[(r + 8 * half) * HB_STRIDE + n] = (bf16_t)h[r];
    if ((t + 1) < Tn) {
      v4f q = *(const v4f*)&xstage[cr * XS_STRIDE + cc];
      v4bf b;
#pragma unroll
      for (int i = 0; i < 4; ++i) b[i] = (bf16_t)q[i];
      *(v4bf*)&xb16[cr * XB_STRIDE + cc] = b;
    }
    __syncthreads();   // new h + new x visible
  }

  // Final h_t, c_t.
  float* hT = out + (size_t)Bn * Tn * Hn;
  float* cT = hT + (size_t)Bn * Hn;
#pragma unroll
  for (int r = 0; r < 8; ++r) {
    int b = b0 + r + 8 * half;
    hT[(size_t)b * Hn + n] = h[r];
    cT[(size_t)b * Hn + n] = c[r];
  }
}

// ---------------------------------------------------------------------------
extern "C" void kernel_launch(void* const* d_in, const int* in_sizes, int n_in,
                              void* d_out, int out_size, void* d_ws,
                              size_t ws_size, hipStream_t stream) {
  (void)in_sizes; (void)n_in; (void)out_size; (void)ws_size;

  const float* x = (const float*)d_in[0];
  const float* U[4] = {(const float*)d_in[1], (const float*)d_in[3],
                       (const float*)d_in[5], (const float*)d_in[7]};  // f,i,c,o
  const float* W[4] = {(const float*)d_in[2], (const float*)d_in[4],
                       (const float*)d_in[6], (const float*)d_in[8]};  // f,i,c,o
  const float* Vf  = (const float*)d_in[9];
  const float* Vi  = (const float*)d_in[10];
  const float* Vo  = (const float*)d_in[11];
  const float* bf_ = (const float*)d_in[12];
  const float* bi_ = (const float*)d_in[13];
  const float* bc_ = (const float*)d_in[14];
  const float* bo_ = (const float*)d_in[15];

  // Workspace: 4x Ut [Hn][In] then 4x Wt [Hn][Hn], bf16 (768 KB total).
  bf16_t* Ut = (bf16_t*)d_ws;
  bf16_t* Wt = Ut + (size_t)4 * Hn * In;

  for (int g = 0; g < 4; ++g) {
    cvt_transpose_kernel<<<(In * Hn + 255) / 256, 256, 0, stream>>>(
        U[g], Ut + (size_t)g * Hn * In, In, Hn);
    cvt_transpose_kernel<<<(Hn * Hn + 255) / 256, 256, 0, stream>>>(
        W[g], Wt + (size_t)g * Hn * Hn, Hn, Hn);
  }

  lstm_peephole_wmma_kernel<<<Bn / 16, 512, 0, stream>>>(
      x, Ut, Wt, Vf, Vi, Vo, bf_, bi_, bc_, bo_, (float*)d_out);
}
